// MoE_50921132261886
// MI455X (gfx1250) — compile-verified, matrix operand order checked by reading, not run on hardware
//
#include <hip/hip_runtime.h>
#include <hip/hip_bf16.h>
#include <math.h>

#define T_TOK 16384
#define DIMSZ 1024
#define HIDSZ 512
#define NEXP  64
#define NROWS (T_TOK * 2)   // T * top_k = 32768
#define TM 128              // rows per tile
#define TN 64               // cols per block
#define KC 64               // K step per stage (2x WMMA K=32)
#define LDA 72              // LDS row stride (bf16 elems), padded (144B, 16B-aligned)
#define ABUF (TM * LDA)     // one A-stage buffer, in elements
#define MAXTILES 320        // >= NROWS/TM + NEXP upper bound

typedef __attribute__((ext_vector_type(16))) __bf16 v16bf;
typedef __attribute__((ext_vector_type(8)))  __bf16 v8bf;
typedef __attribute__((ext_vector_type(8)))  float  v8f;

__device__ __forceinline__ unsigned short f2bfu(float f) {
  union { float f; unsigned u; } v; v.f = f;
  return (unsigned short)((v.u + 0x7FFFu + ((v.u >> 16) & 1u)) >> 16);
}
__device__ __forceinline__ __bf16 ubf(unsigned short h) {
  __bf16 b; __builtin_memcpy(&b, &h, 2); return b;
}

// A fragment (16x32 bf16) from LDS tile (row-major, stride LD).
// Lane L<16: M=L, K chunks {0..7, 16..23}; lane 16+L: M=L, K {8..15, 24..31}.
template <int LD>
__device__ __forceinline__ v16bf lds_a_frag(const unsigned short* sm, int mbase, int ks, int lane) {
  const int half = (lane >> 4) & 1;
  const unsigned short* p = sm + (mbase + (lane & 15)) * LD + ks + half * 8;
  v8bf lo = *(const v8bf*)(p);
  v8bf hi = *(const v8bf*)(p + 16);
  return __builtin_shufflevector(lo, hi, 0,1,2,3,4,5,6,7,8,9,10,11,12,13,14,15);
}

// B fragment (32x16, supplied as NxK row-major bf16 weights): two 16B vector loads.
__device__ __forceinline__ v16bf gl_b_frag_bf16(const unsigned short* w, int ld, int nbase,
                                                int k, int lane) {
  const int half = (lane >> 4) & 1;
  const unsigned short* p = w + (size_t)(nbase + (lane & 15)) * ld + k + half * 8;
  v8bf lo = *(const v8bf*)(p);
  v8bf hi = *(const v8bf*)(p + 16);
  return __builtin_shufflevector(lo, hi, 0,1,2,3,4,5,6,7,8,9,10,11,12,13,14,15);
}

// B fragment from fp32 weights, converted in-register (fallback path).
__device__ __forceinline__ v16bf gl_b_frag_f32(const float* w, int ld, int nbase, int k, int lane) {
  const int half = (lane >> 4) & 1;
  const float* p = w + (size_t)(nbase + (lane & 15)) * ld + k + half * 8;
  v16bf r;
#pragma unroll
  for (int i = 0; i < 8; ++i) {
    r[i]     = ubf(f2bfu(p[i]));
    r[i + 8] = ubf(f2bfu(p[i + 16]));
  }
  return r;
}

#define WMMA_BF16(A, B, C) \
  __builtin_amdgcn_wmma_f32_16x16x32_bf16(false, (A), false, (B), (short)0, (C), false, false)

__global__ void moe_zero_meta(int* counts, int* cursor) {
  int i = threadIdx.x;
  if (i < NEXP) { counts[i] = 0; cursor[i] = 0; }
}

__global__ void moe_zero_out(float* y) {
  size_t i = (size_t)blockIdx.x * blockDim.x + threadIdx.x;
  float4 z = make_float4(0.f, 0.f, 0.f, 0.f);
  ((float4*)y)[i] = z;
}

// fp32 -> bf16 bulk converter (8 elems/thread, 16B stores).
__global__ void moe_cvt_bf16(const float* __restrict__ src, unsigned short* __restrict__ dst,
                             int n8) {
  int i = blockIdx.x * blockDim.x + threadIdx.x;
  if (i >= n8) return;
  const float4 a = ((const float4*)src)[(size_t)i * 2];
  const float4 b = ((const float4*)src)[(size_t)i * 2 + 1];
  union { unsigned short s[8]; uint4 u; } o;
  o.s[0] = f2bfu(a.x); o.s[1] = f2bfu(a.y); o.s[2] = f2bfu(a.z); o.s[3] = f2bfu(a.w);
  o.s[4] = f2bfu(b.x); o.s[5] = f2bfu(b.y); o.s[6] = f2bfu(b.z); o.s[7] = f2bfu(b.w);
  ((uint4*)dst)[i] = o.u;
}

// One block per token, 64 threads (one per expert): scores, softmax, top-2, histogram.
__global__ void moe_router(const float* __restrict__ x, const float* __restrict__ gw,
                           int* __restrict__ topk_idx, float* __restrict__ topk_score,
                           int* __restrict__ counts) {
  __shared__ float sc[NEXP];
  const int t = blockIdx.x, e = threadIdx.x;
  const float* xr = x + (size_t)t * DIMSZ;
  const float* wr = gw + (size_t)e * DIMSZ;
  float acc = 0.f;
  for (int k = 0; k < DIMSZ; k += 4) {
    float4 xv = *(const float4*)(xr + k);
    float4 wv = *(const float4*)(wr + k);
    acc += xv.x * wv.x + xv.y * wv.y + xv.z * wv.z + xv.w * wv.w;
  }
  sc[e] = acc;
  __syncthreads();
  if (e == 0) {
    float mx = sc[0];
    for (int i = 1; i < NEXP; ++i) mx = fmaxf(mx, sc[i]);
    float sum = 0.f;
    for (int i = 0; i < NEXP; ++i) { sc[i] = __expf(sc[i] - mx); sum += sc[i]; }
    const float inv = 1.f / sum;
    int i0 = 0; float v0 = -1.f;
    for (int i = 0; i < NEXP; ++i) if (sc[i] > v0) { v0 = sc[i]; i0 = i; }
    int i1 = 0; float v1 = -1.f;
    for (int i = 0; i < NEXP; ++i) if (i != i0 && sc[i] > v1) { v1 = sc[i]; i1 = i; }
    topk_idx[t * 2] = i0;        topk_idx[t * 2 + 1] = i1;
    topk_score[t * 2] = v0 * inv; topk_score[t * 2 + 1] = v1 * inv;
    atomicAdd(&counts[i0], 1);
    atomicAdd(&counts[i1], 1);
  }
}

__global__ void moe_scan(const int* __restrict__ counts, int* __restrict__ offsets,
                         int* __restrict__ numTiles, int* __restrict__ worklist) {
  if (threadIdx.x == 0 && blockIdx.x == 0) {
    int off = 0;
    for (int e = 0; e < NEXP; ++e) { offsets[e] = off; off += counts[e]; }
    offsets[NEXP] = off;
    int nt = 0;
    for (int e = 0; e < NEXP; ++e) {
      const int tiles = (counts[e] + TM - 1) / TM;
      for (int m = 0; m < tiles; ++m) worklist[nt++] = (e << 16) | m;
    }
    numTiles[0] = nt;
  }
}

__global__ void moe_scatter(const int* __restrict__ topk_idx, const float* __restrict__ topk_score,
                            const int* __restrict__ offsets, int* __restrict__ cursor,
                            int* __restrict__ row_tok, float* __restrict__ row_score) {
  int i = blockIdx.x * blockDim.x + threadIdx.x;
  if (i >= NROWS) return;
  const int e = topk_idx[i];
  const int pos = offsets[e] + atomicAdd(&cursor[e], 1);
  row_tok[pos] = i >> 1;
  row_score[pos] = topk_score[i];
}

// ---------------- Fast path: all-bf16 operands, double-buffered LDS ----------------

// GEMM1: h = silu(R @ W1^T) * (R @ W3^T). A from xb (bf16), B from w1b/w3b (bf16).
__global__ __launch_bounds__(256) void moe_ffn1_bf16(
    const unsigned short* __restrict__ xb, const unsigned short* __restrict__ w1b,
    const unsigned short* __restrict__ w3b,
    const int* __restrict__ offsets, const int* __restrict__ numTiles,
    const int* __restrict__ worklist, const int* __restrict__ row_tok,
    unsigned short* __restrict__ hbuf) {
  __shared__ unsigned short smA[2 * ABUF];   // double-buffered A tile (offset-selected)
  __shared__ int smTok[TM];
  const int nt = numTiles[0];
  if ((int)blockIdx.x >= nt) return;
  const int wle = worklist[blockIdx.x];
  const int e = wle >> 16, mt = wle & 0xFFFF;
  const int beg = offsets[e];
  const int n_e = offsets[e + 1] - beg;
  const int base = beg + mt * TM;
  const int rows = min(TM, n_e - mt * TM);
  const int tid = threadIdx.x;
  if (tid < TM) smTok[tid] = row_tok[base + min(tid, rows - 1)];
  __syncthreads();

  const int lane = tid & 31, wave = tid >> 5;
  const int wm = wave & 3, wn = wave >> 2;
  const int nb = blockIdx.y * TN + wn * 32;
  const unsigned short* w1e = w1b + (size_t)e * HIDSZ * DIMSZ;
  const unsigned short* w3e = w3b + (size_t)e * HIDSZ * DIMSZ;

  v8f acc1[2][2] = {}; v8f acc3[2][2] = {};

  const int arow = tid >> 1, acb = (tid & 1) * 32;  // 32 bf16 per thread per stage
  const int abase = arow * LDA + acb;               // per-thread staging offset in a buffer
  const unsigned short* asrc = xb + (size_t)smTok[arow] * DIMSZ + acb;

  { // prologue: stage tile 0 into buffer 0 (pure LDS indexing -> ds_store_b128)
    uint4 u0 = *(const uint4*)(asrc);
    uint4 u1 = *(const uint4*)(asrc + 8);
    uint4 u2 = *(const uint4*)(asrc + 16);
    uint4 u3 = *(const uint4*)(asrc + 24);
    uint4* d = (uint4*)(smA + abase);
    d[0] = u0; d[1] = u1; d[2] = u2; d[3] = u3;
  }
  __syncthreads();

  for (int kk = 0; kk < DIMSZ; kk += KC) {
    const int cur = (kk >> 6) & 1;
    const unsigned short* smc = smA + cur * ABUF;
    const bool more = (kk + KC < DIMSZ);
    uint4 n0, n1, n2, n3;
    if (more) { // next A stage -> registers (overlaps with WMMA chain)
      n0 = *(const uint4*)(asrc + kk + KC);
      n1 = *(const uint4*)(asrc + kk + KC + 8);
      n2 = *(const uint4*)(asrc + kk + KC + 16);
      n3 = *(const uint4*)(asrc + kk + KC + 24);
      __builtin_prefetch(w1e + (size_t)nb * DIMSZ + kk + KC, 0, 1);
      __builtin_prefetch(w3e + (size_t)nb * DIMSZ + kk + KC, 0, 1);
    }
    // hoist all fragment loads for this K-step ahead of the WMMA chain
    v16bf a[2][2], b1[2][2], b3[2][2];     // [ks][m] / [ks][ns]
#pragma unroll
    for (int ki = 0; ki < 2; ++ki) {
      a[ki][0] = lds_a_frag<LDA>(smc, wm * 32, ki * 32, lane);
      a[ki][1] = lds_a_frag<LDA>(smc, wm * 32 + 16, ki * 32, lane);
#pragma unroll
      for (int ns = 0; ns < 2; ++ns) {
        b1[ki][ns] = gl_b_frag_bf16(w1e, DIMSZ, nb + ns * 16, kk + ki * 32, lane);
        b3[ki][ns] = gl_b_frag_bf16(w3e, DIMSZ, nb + ns * 16, kk + ki * 32, lane);
      }
    }
#pragma unroll
    for (int ki = 0; ki < 2; ++ki)
#pragma unroll
      for (int ns = 0; ns < 2; ++ns) {
        acc1[0][ns] = WMMA_BF16(a[ki][0], b1[ki][ns], acc1[0][ns]);
        acc1[1][ns] = WMMA_BF16(a[ki][1], b1[ki][ns], acc1[1][ns]);
        acc3[0][ns] = WMMA_BF16(a[ki][0], b3[ki][ns], acc3[0][ns]);
        acc3[1][ns] = WMMA_BF16(a[ki][1], b3[ki][ns], acc3[1][ns]);
      }
    if (more) { // write next stage to the other buffer (offset select -> ds_store_b128)
      uint4* d = (uint4*)(smA + (1 - cur) * ABUF + abase);
      d[0] = n0; d[1] = n1; d[2] = n2; d[3] = n3;
    }
    __syncthreads();
  }

#pragma unroll
  for (int ms = 0; ms < 2; ++ms)
#pragma unroll
    for (int ns = 0; ns < 2; ++ns)
#pragma unroll
      for (int i = 0; i < 8; ++i) {
        const int lm = wm * 32 + ms * 16 + ((lane < 16) ? i : i + 8);
        const int col = nb + ns * 16 + (lane & 15);
        if (lm < rows) {
          const float s = acc1[ms][ns][i];
          const float g = s / (1.f + __expf(-s));
          hbuf[(size_t)(base + lm) * HIDSZ + col] = f2bfu(g * acc3[ms][ns][i]);
        }
      }
}

// GEMM2: out = h @ W2^T (bf16 x bf16), score-scaled atomicAdd into y.
__global__ __launch_bounds__(256) void moe_ffn2_bf16(
    const unsigned short* __restrict__ hbuf, const unsigned short* __restrict__ w2b,
    const int* __restrict__ offsets, const int* __restrict__ numTiles,
    const int* __restrict__ worklist, const int* __restrict__ row_tok,
    const float* __restrict__ row_score, float* __restrict__ y) {
  __shared__ unsigned short smA[2 * ABUF];
  __shared__ int smTok[TM];
  __shared__ float smSc[TM];
  const int nt = numTiles[0];
  if ((int)blockIdx.x >= nt) return;
  const int wle = worklist[blockIdx.x];
  const int e = wle >> 16, mt = wle & 0xFFFF;
  const int beg = offsets[e];
  const int n_e = offsets[e + 1] - beg;
  const int base = beg + mt * TM;
  const int rows = min(TM, n_e - mt * TM);
  const int tid = threadIdx.x;
  if (tid < TM) {
    const int r = min(tid, rows - 1);
    smTok[tid] = row_tok[base + r];
    smSc[tid]  = row_score[base + r];
  }
  __syncthreads();

  const int lane = tid & 31, wave = tid >> 5;
  const int wm = wave & 3, wn = wave >> 2;
  const int nb = blockIdx.y * TN + wn * 32;
  const unsigned short* w2e = w2b + (size_t)e * DIMSZ * HIDSZ;

  v8f acc[2][2] = {};

  const int arow = tid >> 1, acb = (tid & 1) * 32;
  const int abase = arow * LDA + acb;
  const unsigned short* asrc = hbuf + (size_t)(base + min(arow, rows - 1)) * HIDSZ + acb;

  {
    uint4 u0 = *(const uint4*)(asrc);
    uint4 u1 = *(const uint4*)(asrc + 8);
    uint4 u2 = *(const uint4*)(asrc + 16);
    uint4 u3 = *(const uint4*)(asrc + 24);
    uint4* d = (uint4*)(smA + abase);
    d[0] = u0; d[1] = u1; d[2] = u2; d[3] = u3;
  }
  __syncthreads();

  for (int kk = 0; kk < HIDSZ; kk += KC) {
    const int cur = (kk >> 6) & 1;
    const unsigned short* smc = smA + cur * ABUF;
    const bool more = (kk + KC < HIDSZ);
    uint4 n0, n1, n2, n3;
    if (more) {
      n0 = *(const uint4*)(asrc + kk + KC);
      n1 = *(const uint4*)(asrc + kk + KC + 8);
      n2 = *(const uint4*)(asrc + kk + KC + 16);
      n3 = *(const uint4*)(asrc + kk + KC + 24);
      __builtin_prefetch(w2e + (size_t)nb * HIDSZ + kk + KC, 0, 1);
    }
    v16bf a[2][2], b[2][2];
#pragma unroll
    for (int ki = 0; ki < 2; ++ki) {
      a[ki][0] = lds_a_frag<LDA>(smc, wm * 32, ki * 32, lane);
      a[ki][1] = lds_a_frag<LDA>(smc, wm * 32 + 16, ki * 32, lane);
#pragma unroll
      for (int ns = 0; ns < 2; ++ns)
        b[ki][ns] = gl_b_frag_bf16(w2e, HIDSZ, nb + ns * 16, kk + ki * 32, lane);
    }
#pragma unroll
    for (int ki = 0; ki < 2; ++ki)
#pragma unroll
      for (int ns = 0; ns < 2; ++ns) {
        acc[0][ns] = WMMA_BF16(a[ki][0], b[ki][ns], acc[0][ns]);
        acc[1][ns] = WMMA_BF16(a[ki][1], b[ki][ns], acc[1][ns]);
      }
    if (more) {
      uint4* d = (uint4*)(smA + (1 - cur) * ABUF + abase);
      d[0] = n0; d[1] = n1; d[2] = n2; d[3] = n3;
    }
    __syncthreads();
  }

#pragma unroll
  for (int ms = 0; ms < 2; ++ms)
#pragma unroll
    for (int ns = 0; ns < 2; ++ns)
#pragma unroll
      for (int i = 0; i < 8; ++i) {
        const int lm = wm * 32 + ms * 16 + ((lane < 16) ? i : i + 8);
        const int col = nb + ns * 16 + (lane & 15);
        if (lm < rows) {
          atomicAdd(&y[(size_t)smTok[lm] * DIMSZ + col], acc[ms][ns][i] * smSc[lm]);
        }
      }
}

// ---------------- Fallback path (small ws): fp32 weights converted in-register ----------------

__global__ __launch_bounds__(256) void moe_ffn1_f32(
    const float* __restrict__ x, const float* __restrict__ w1, const float* __restrict__ w3,
    const int* __restrict__ offsets, const int* __restrict__ numTiles,
    const int* __restrict__ worklist, const int* __restrict__ row_tok,
    unsigned short* __restrict__ hbuf) {
  __shared__ unsigned short smA[TM * 40];
  __shared__ int smTok[TM];
  const int nt = numTiles[0];
  if ((int)blockIdx.x >= nt) return;
  const int wle = worklist[blockIdx.x];
  const int e = wle >> 16, mt = wle & 0xFFFF;
  const int beg = offsets[e];
  const int n_e = offsets[e + 1] - beg;
  const int base = beg + mt * TM;
  const int rows = min(TM, n_e - mt * TM);
  const int tid = threadIdx.x;
  if (tid < TM) smTok[tid] = row_tok[base + min(tid, rows - 1)];
  __syncthreads();

  const int lane = tid & 31, wave = tid >> 5;
  const int wm = wave & 3, wn = wave >> 2;
  const int nb = blockIdx.y * TN + wn * 32;
  const float* w1e = w1 + (size_t)e * HIDSZ * DIMSZ;
  const float* w3e = w3 + (size_t)e * HIDSZ * DIMSZ;

  v8f acc1[2][2] = {}; v8f acc3[2][2] = {};

  const int arow = tid >> 1, acb = (tid & 1) * 16;
  const float* asrc = x + (size_t)smTok[arow] * DIMSZ + acb;

  for (int kk = 0; kk < DIMSZ; kk += 32) {
    {
      float4 f0 = *(const float4*)(asrc + kk);
      float4 f1 = *(const float4*)(asrc + kk + 4);
      float4 f2 = *(const float4*)(asrc + kk + 8);
      float4 f3 = *(const float4*)(asrc + kk + 12);
      unsigned short* d = smA + arow * 40 + acb;
      d[0]=f2bfu(f0.x); d[1]=f2bfu(f0.y); d[2]=f2bfu(f0.z); d[3]=f2bfu(f0.w);
      d[4]=f2bfu(f1.x); d[5]=f2bfu(f1.y); d[6]=f2bfu(f1.z); d[7]=f2bfu(f1.w);
      d[8]=f2bfu(f2.x); d[9]=f2bfu(f2.y); d[10]=f2bfu(f2.z); d[11]=f2bfu(f2.w);
      d[12]=f2bfu(f3.x); d[13]=f2bfu(f3.y); d[14]=f2bfu(f3.z); d[15]=f2bfu(f3.w);
    }
    __syncthreads();
    v16bf a0 = lds_a_frag<40>(smA, wm * 32, 0, lane);
    v16bf a1 = lds_a_frag<40>(smA, wm * 32 + 16, 0, lane);
#pragma unroll
    for (int ns = 0; ns < 2; ++ns) {
      v16bf b1 = gl_b_frag_f32(w1e, DIMSZ, nb + ns * 16, kk, lane);
      v16bf b3 = gl_b_frag_f32(w3e, DIMSZ, nb + ns * 16, kk, lane);
      acc1[0][ns] = WMMA_BF16(a0, b1, acc1[0][ns]);
      acc1[1][ns] = WMMA_BF16(a1, b1, acc1[1][ns]);
      acc3[0][ns] = WMMA_BF16(a0, b3, acc3[0][ns]);
      acc3[1][ns] = WMMA_BF16(a1, b3, acc3[1][ns]);
    }
    __syncthreads();
  }

#pragma unroll
  for (int ms = 0; ms < 2; ++ms)
#pragma unroll
    for (int ns = 0; ns < 2; ++ns)
#pragma unroll
      for (int i = 0; i < 8; ++i) {
        const int lm = wm * 32 + ms * 16 + ((lane < 16) ? i : i + 8);
        const int col = nb + ns * 16 + (lane & 15);
        if (lm < rows) {
          const float s = acc1[ms][ns][i];
          const float g = s / (1.f + __expf(-s));
          hbuf[(size_t)(base + lm) * HIDSZ + col] = f2bfu(g * acc3[ms][ns][i]);
        }
      }
}

__global__ __launch_bounds__(256) void moe_ffn2_f32(
    const unsigned short* __restrict__ hbuf, const float* __restrict__ w2,
    const int* __restrict__ offsets, const int* __restrict__ numTiles,
    const int* __restrict__ worklist, const int* __restrict__ row_tok,
    const float* __restrict__ row_score, float* __restrict__ y) {
  __shared__ unsigned short smA[TM * 40];
  __shared__ int smTok[TM];
  __shared__ float smSc[TM];
  const int nt = numTiles[0];
  if ((int)blockIdx.x >= nt) return;
  const int wle = worklist[blockIdx.x];
  const int e = wle >> 16, mt = wle & 0xFFFF;
  const int beg = offsets[e];
  const int n_e = offsets[e + 1] - beg;
  const int base = beg + mt * TM;
  const int rows = min(TM, n_e - mt * TM);
  const int tid = threadIdx.x;
  if (tid < TM) {
    const int r = min(tid, rows - 1);
    smTok[tid] = row_tok[base + r];
    smSc[tid]  = row_score[base + r];
  }
  __syncthreads();

  const int lane = tid & 31, wave = tid >> 5;
  const int wm = wave & 3, wn = wave >> 2;
  const int nb = blockIdx.y * TN + wn * 32;
  const float* w2e = w2 + (size_t)e * DIMSZ * HIDSZ;

  v8f acc[2][2] = {};

  const int arow = tid >> 1, acb = (tid & 1) * 16;
  const unsigned short* asrc = hbuf + (size_t)(base + min(arow, rows - 1)) * HIDSZ + acb;

  for (int kk = 0; kk < HIDSZ; kk += 32) {
    {
      uint4 u0 = *(const uint4*)(asrc + kk);
      uint4 u1 = *(const uint4*)(asrc + kk + 8);
      *(uint4*)(smA + arow * 40 + acb) = u0;
      *(uint4*)(smA + arow * 40 + acb + 8) = u1;
    }
    __syncthreads();
    v16bf a0 = lds_a_frag<40>(smA, wm * 32, 0, lane);
    v16bf a1 = lds_a_frag<40>(smA, wm * 32 + 16, 0, lane);
#pragma unroll
    for (int ns = 0; ns < 2; ++ns) {
      v16bf b = gl_b_frag_f32(w2e, HIDSZ, nb + ns * 16, kk, lane);
      acc[0][ns] = WMMA_BF16(a0, b, acc[0][ns]);
      acc[1][ns] = WMMA_BF16(a1, b, acc[1][ns]);
    }
    __syncthreads();
  }

#pragma unroll
  for (int ms = 0; ms < 2; ++ms)
#pragma unroll
    for (int ns = 0; ns < 2; ++ns)
#pragma unroll
      for (int i = 0; i < 8; ++i) {
        const int lm = wm * 32 + ms * 16 + ((lane < 16) ? i : i + 8);
        const int col = nb + ns * 16 + (lane & 15);
        if (lm < rows) {
          atomicAdd(&y[(size_t)smTok[lm] * DIMSZ + col], acc[ms][ns][i] * smSc[lm]);
        }
      }
}

extern "C" void kernel_launch(void* const* d_in, const int* in_sizes, int n_in,
                              void* d_out, int out_size, void* d_ws, size_t ws_size,
                              hipStream_t stream) {
  const float* x  = (const float*)d_in[0];
  const float* gw = (const float*)d_in[1];
  const float* w1 = (const float*)d_in[2];
  const float* w2 = (const float*)d_in[3];
  const float* w3 = (const float*)d_in[4];
  float* y = (float*)d_out;

  int* ws = (int*)d_ws;
  int* counts      = ws;                // 64
  int* cursor      = ws + 64;           // 64
  int* offsets     = ws + 128;          // 65
  int* numTiles    = ws + 224;          // 1
  int* worklist    = ws + 256;          // < 512
  int* topk_idx    = ws + 1024;                         // NROWS
  float* topk_score = (float*)(ws + 1024 + NROWS);      // NROWS
  int* row_tok     = ws + 1024 + 2 * NROWS;             // NROWS
  float* row_score = (float*)(ws + 1024 + 3 * NROWS);   // NROWS
  unsigned short* hbuf = (unsigned short*)(ws + 1024 + 4 * NROWS);  // NROWS*HID
  unsigned short* xb  = hbuf + (size_t)NROWS * HIDSZ;               // T*DIM
  unsigned short* w1b = xb  + (size_t)T_TOK * DIMSZ;                // E*HID*DIM
  unsigned short* w2b = w1b + (size_t)NEXP * HIDSZ * DIMSZ;         // E*DIM*HID
  unsigned short* w3b = w2b + (size_t)NEXP * DIMSZ * HIDSZ;         // E*HID*DIM
  const size_t need_big =
      (size_t)((char*)(w3b + (size_t)NEXP * HIDSZ * DIMSZ) - (char*)d_ws);

  moe_zero_meta<<<1, 128, 0, stream>>>(counts, cursor);
  moe_zero_out<<<(T_TOK * DIMSZ / 4 + 255) / 256, 256, 0, stream>>>(y);
  moe_router<<<T_TOK, NEXP, 0, stream>>>(x, gw, topk_idx, topk_score, counts);
  moe_scan<<<1, 64, 0, stream>>>(counts, offsets, numTiles, worklist);
  moe_scatter<<<(NROWS + 255) / 256, 256, 0, stream>>>(topk_idx, topk_score, offsets, cursor,
                                                       row_tok, row_score);

  if (ws_size >= need_big) {
    const int nx = T_TOK * DIMSZ / 8;           // 2,097,152
    const int nw = NEXP * HIDSZ * DIMSZ / 8;    // 4,194,304
    moe_cvt_bf16<<<(nx + 255) / 256, 256, 0, stream>>>(x, xb, nx);
    moe_cvt_bf16<<<(nw + 255) / 256, 256, 0, stream>>>(w1, w1b, nw);
    moe_cvt_bf16<<<(nw + 255) / 256, 256, 0, stream>>>(w2, w2b, nw);
    moe_cvt_bf16<<<(nw + 255) / 256, 256, 0, stream>>>(w3, w3b, nw);
    moe_ffn1_bf16<<<dim3(MAXTILES, HIDSZ / TN), 256, 0, stream>>>(
        xb, w1b, w3b, offsets, numTiles, worklist, row_tok, hbuf);
    moe_ffn2_bf16<<<dim3(MAXTILES, DIMSZ / TN), 256, 0, stream>>>(
        hbuf, w2b, offsets, numTiles, worklist, row_tok, row_score, y);
  } else {
    moe_ffn1_f32<<<dim3(MAXTILES, HIDSZ / TN), 256, 0, stream>>>(
        x, w1, w3, offsets, numTiles, worklist, row_tok, hbuf);
    moe_ffn2_f32<<<dim3(MAXTILES, DIMSZ / TN), 256, 0, stream>>>(
        hbuf, w2, offsets, numTiles, worklist, row_tok, row_score, y);
  }
}
